// VQVAEStressAdapter_69432441307337
// MI455X (gfx1250) — matrix operand order repaired and checked by reading, not compiled, not run on hardware
//
#include <hip/hip_runtime.h>
#include <hip/hip_bf16.h>
#include <math.h>

// ---------------------------------------------------------------------------
// Types for CDNA5 WMMA (wave32, v_wmma_f32_16x16x32_bf16)
// ---------------------------------------------------------------------------
typedef unsigned short u16b;
typedef __attribute__((ext_vector_type(16))) __bf16 v16bf;
typedef __attribute__((ext_vector_type(8)))  float  v8f;

union FragU { uint4 u[2]; v16bf v; };

__device__ __forceinline__ u16b f2bf(float f) {
  unsigned u = __float_as_uint(f);
  unsigned r = 0x7FFFu + ((u >> 16) & 1u);
  return (u16b)((u + r) >> 16);
}

// ---------------------------------------------------------------------------
// im2col: X[b][ci][Lin] (float) -> dst[n][Kpad] (bf16), n = b*Lout + l
// column kk = ci*kw + t maps to input position l*stride + t - pad (OOB -> 0).
// Optional ReLU applied to the *input* (fuses the res-stack ReLUs).
// ---------------------------------------------------------------------------
__global__ void k_im2col(const float* __restrict__ X, u16b* __restrict__ dst,
                         int Cin, int Lin, int Lout, int kw, int stride, int pad,
                         int Kpad, int relu_in)
{
  long long idx = (long long)blockIdx.x * blockDim.x + threadIdx.x;
  long long total = (long long)128 * Lout * Kpad;
  if (idx >= total) return;
  int kk = (int)(idx % Kpad);
  long long n = idx / Kpad;
  float v = 0.f;
  int Ktot = Cin * kw;
  if (kk < Ktot) {
    int ci = kk / kw, t = kk % kw;
    int b = (int)(n / Lout), l = (int)(n % Lout);
    int pos = l * stride + t - pad;
    if (pos >= 0 && pos < Lin) {
      v = X[((long long)b * Cin + ci) * Lin + pos];
      if (relu_in) v = fmaxf(v, 0.f);
    }
  }
  dst[idx] = f2bf(v);
}

// Weight prep (normal conv): src [Cout][Cin][kw] -> dst [Mpad][Kpad] bf16
__global__ void k_prep_w(const float* __restrict__ W, u16b* __restrict__ dst,
                         int Cout, int Cin, int kw, int Kpad, int Mpad)
{
  int idx = blockIdx.x * blockDim.x + threadIdx.x;
  if (idx >= Mpad * Kpad) return;
  int co = idx / Kpad, kk = idx % Kpad;
  float v = 0.f;
  if (co < Cout && kk < Cin * kw) v = W[(long long)co * Cin * kw + kk];
  dst[idx] = f2bf(v);
}

// Weight prep (transpose conv parity): src [Cin][Cout][kw];
// dst[co][ci*2+u] = W[ci][co][tap_u]  (kw=2 effective taps per parity)
__global__ void k_prep_wt(const float* __restrict__ W, u16b* __restrict__ dst,
                          int Cout, int Cin, int kw, int Kpad, int Mpad,
                          int tap0, int tap1)
{
  int idx = blockIdx.x * blockDim.x + threadIdx.x;
  if (idx >= Mpad * Kpad) return;
  int co = idx / Kpad, kk = idx % Kpad;
  float v = 0.f;
  if (co < Cout && kk < Cin * 2) {
    int ci = kk >> 1, u = kk & 1;
    int t = u ? tap1 : tap0;
    v = W[((long long)ci * Cout + co) * kw + t];
  }
  dst[idx] = f2bf(v);
}

// ---------------------------------------------------------------------------
// WMMA GEMM, 2x2 tiles per wave (32M x 32N): D = A*B, A = weights
// [Mpad][Kpad] bf16, B K-major per column: Bm[n][Kpad] bf16. 8 waves/block
// -> block covers 32M x 256N. 4 v_wmma per 4 fragment loads (16 FLOP/B).
// Epilogue: +bias, +residual, ReLU, scatter store
// out[b][m][ l*lscale + loff ] with row stride Lstore  (n = b*Lout + l).
// ---------------------------------------------------------------------------
__global__ void __launch_bounds__(256)
k_gemm(const u16b* __restrict__ A, const u16b* __restrict__ Bm,
       const float* __restrict__ bias, const float* __restrict__ resid,
       float* __restrict__ out,
       int Mout, int Kpad, int Lout, int Lstore, int lscale, int loff, int relu)
{
  const int wave = threadIdx.x >> 5;
  const int lane = threadIdx.x & 31;
  const int hh   = lane >> 4;      // half-wave select
  const int lm   = lane & 15;
  const int n0 = (blockIdx.x * 8 + wave) * 32;
  const int m0 = blockIdx.y * 32;

  // A frags: rows m0+lm, m0+16+lm; K runs [kb..kb+7],[kb+16..kb+23], kb=0/8
  const u16b* arow0 = A  + (long long)(m0 + lm)      * Kpad + (hh ? 8 : 0);
  const u16b* arow1 = A  + (long long)(m0 + 16 + lm) * Kpad + (hh ? 8 : 0);
  // B frags: columns n0+lm, n0+16+lm; K runs [kb..kb+7],[kb+16..kb+23], kb=0/8
  const u16b* brow0 = Bm + (long long)(n0 + lm)      * Kpad + (hh ? 8 : 0);
  const u16b* brow1 = Bm + (long long)(n0 + 16 + lm) * Kpad + (hh ? 8 : 0);

  v8f acc00 = {0.f,0.f,0.f,0.f,0.f,0.f,0.f,0.f};
  v8f acc01 = acc00, acc10 = acc00, acc11 = acc00;

  const int ksteps = Kpad >> 5;
  for (int ks = 0; ks < ksteps; ++ks) {
    FragU fa0, fa1, fb0, fb1;
    fa0.u[0] = *(const uint4*)(arow0);
    fa0.u[1] = *(const uint4*)(arow0 + 16);
    fa1.u[0] = *(const uint4*)(arow1);
    fa1.u[1] = *(const uint4*)(arow1 + 16);
    fb0.u[0] = *(const uint4*)(brow0);
    fb0.u[1] = *(const uint4*)(brow0 + 16);
    fb1.u[0] = *(const uint4*)(brow1);
    fb1.u[1] = *(const uint4*)(brow1 + 16);
    __builtin_prefetch((const void*)(brow0 + 32), 0, 0);
    __builtin_prefetch((const void*)(brow1 + 32), 0, 0);
    arow0 += 32; arow1 += 32; brow0 += 32; brow1 += 32;
    acc00 = __builtin_amdgcn_wmma_f32_16x16x32_bf16(
        false, fa0.v, false, fb0.v, (short)0, acc00, false, false);
    acc01 = __builtin_amdgcn_wmma_f32_16x16x32_bf16(
        false, fa0.v, false, fb1.v, (short)0, acc01, false, false);
    acc10 = __builtin_amdgcn_wmma_f32_16x16x32_bf16(
        false, fa1.v, false, fb0.v, (short)0, acc10, false, false);
    acc11 = __builtin_amdgcn_wmma_f32_16x16x32_bf16(
        false, fa1.v, false, fb1.v, (short)0, acc11, false, false);
  }

  const bool hasB = (bias != nullptr), hasR = (resid != nullptr);
#pragma unroll
  for (int tj = 0; tj < 2; ++tj) {
    const int n_g = n0 + tj * 16 + lm;
    const int b = n_g / Lout, l = n_g % Lout;
    const long long obase =
        (long long)b * Mout * Lstore + (long long)l * lscale + loff;
#pragma unroll
    for (int ti = 0; ti < 2; ++ti) {
      const v8f& acc = (tj == 0) ? (ti == 0 ? acc00 : acc10)
                                 : (ti == 0 ? acc01 : acc11);
#pragma unroll
      for (int r = 0; r < 8; ++r) {
        int m = m0 + ti * 16 + r + (hh ? 8 : 0);
        if (m < Mout) {
          float v = acc[r];
          if (hasB) v += bias[m];
          long long oi = obase + (long long)m * Lstore;
          if (hasR) v += resid[oi];
          if (relu) v = fmaxf(v, 0.f);
          out[oi] = v;
        }
      }
    }
  }
}

// cond[b][d] = tanh(classes[b]·cp_w[d] + cp_b[d]);  B=128, D=32, COND=64
__global__ void k_cond(const float* __restrict__ classes,
                       const float* __restrict__ cpw,
                       const float* __restrict__ cpb, float* __restrict__ cond)
{
  int idx = blockIdx.x * blockDim.x + threadIdx.x;
  if (idx >= 128 * 32) return;
  int b = idx >> 5, d = idx & 31;
  float s = cpb[d];
#pragma unroll 4
  for (int c = 0; c < 64; ++c) s = fmaf(classes[b * 64 + c], cpw[d * 64 + c], s);
  cond[idx] = tanhf(s);
}

// VQ: per position n=(b,l): ze = Z[b][:][l] + cond[b][:]; argmin over 256
// codes of ||ze-e||^2; Q = gather; accumulate min dist (== commit numerator).
__global__ void k_vq(const float* __restrict__ Z, const float* __restrict__ cond,
                     const float* __restrict__ cb, float* __restrict__ Q,
                     float* __restrict__ accum, int L)
{
  int n = blockIdx.x * blockDim.x + threadIdx.x;
  float best = 0.f;
  if (n < 128 * L) {
    int b = n / L, l = n % L;
    float ze[32];
#pragma unroll
    for (int d = 0; d < 32; ++d)
      ze[d] = Z[((long long)b * 32 + d) * L + l] + cond[b * 32 + d];
    best = 3.4e38f; int bi = 0;
    for (int e = 0; e < 256; ++e) {
      float s = 0.f;
#pragma unroll
      for (int d = 0; d < 32; ++d) {
        float df = ze[d] - cb[e * 32 + d];
        s = fmaf(df, df, s);
      }
      if (s < best) { best = s; bi = e; }
    }
#pragma unroll
    for (int d = 0; d < 32; ++d)
      Q[((long long)b * 32 + d) * L + l] = cb[bi * 32 + d];
  }
  __shared__ float sm[256];
  sm[threadIdx.x] = best;
  __syncthreads();
  for (int st = 128; st > 0; st >>= 1) {
    if ((int)threadIdx.x < st) sm[threadIdx.x] += sm[threadIdx.x + st];
    __syncthreads();
  }
  if (threadIdx.x == 0) atomicAdd(&accum[1], sm[0]);
}

__global__ void k_recloss(const float* __restrict__ r, const float* __restrict__ x,
                          float* __restrict__ accum, int n)
{
  float s = 0.f;
  for (int i = blockIdx.x * blockDim.x + threadIdx.x; i < n;
       i += gridDim.x * blockDim.x) {
    float d = r[i] - x[i];
    s = fmaf(d, d, s);
  }
  __shared__ float sm[256];
  sm[threadIdx.x] = s;
  __syncthreads();
  for (int st = 128; st > 0; st >>= 1) {
    if ((int)threadIdx.x < st) sm[threadIdx.x] += sm[threadIdx.x + st];
    __syncthreads();
  }
  if (threadIdx.x == 0) atomicAdd(&accum[0], sm[0]);
}

__global__ void k_init(float* a) { a[0] = 0.f; a[1] = 0.f; }

__global__ void k_final(const float* __restrict__ a, float* __restrict__ out)
{
  // rec mean over 128*2112; commit = 0.25 * mean over 128*528*32
  out[0] = a[0] * (1.f / 270336.f) + 0.25f * a[1] * (1.f / 2162688.f);
}

// ---------------------------------------------------------------------------
extern "C" void kernel_launch(void* const* d_in, const int* in_sizes, int n_in,
                              void* d_out, int out_size, void* d_ws, size_t ws_size,
                              hipStream_t stream)
{
  (void)in_sizes; (void)n_in; (void)out_size; (void)ws_size;
  const float* img      = (const float*)d_in[0];
  const float* classes  = (const float*)d_in[1];
  const float* enc1_w   = (const float*)d_in[2];
  const float* enc1_b   = (const float*)d_in[3];
  const float* enc2_w   = (const float*)d_in[4];
  const float* enc2_b   = (const float*)d_in[5];
  const float* enc3_w   = (const float*)d_in[6];
  const float* enc3_b   = (const float*)d_in[7];
  const float* enc_w3   = (const float*)d_in[8];
  const float* enc_w1   = (const float*)d_in[9];
  const float* prevq_w  = (const float*)d_in[10];
  const float* prevq_b  = (const float*)d_in[11];
  const float* cp_w     = (const float*)d_in[12];
  const float* cp_b     = (const float*)d_in[13];
  const float* codebook = (const float*)d_in[14];
  const float* dec1_w   = (const float*)d_in[15];
  const float* dec1_b   = (const float*)d_in[16];
  const float* dec_w3   = (const float*)d_in[17];
  const float* dec_w1   = (const float*)d_in[18];
  const float* ct1_w    = (const float*)d_in[19];
  const float* ct1_b    = (const float*)d_in[20];
  const float* ct2_w    = (const float*)d_in[21];
  const float* ct2_b    = (const float*)d_in[22];

  size_t off = 0;
  auto walloc = [&](size_t bytes) -> void* {
    void* p = (char*)d_ws + off;
    off += (bytes + 255) & ~(size_t)255;
    return p;
  };
  float* F0  = (float*)walloc(34603008);   // [128,128,528] / [128,64,1056]
  float* F1  = (float*)walloc(34603008);
  u16b*  IM  = (u16b*) walloc(51904512);   // im2col, max N*Kpad = 67584*384
  u16b*  WB  = (u16b*) walloc(131072);     // bf16 weights, max 128*384
  float* QB  = (float*)walloc(8650752);    // [128,32,528]
  float* CND = (float*)walloc(16384);      // [128,32]
  float* RC  = (float*)walloc(1081344);    // [128,1,2112]
  float* ACC = (float*)walloc(256);        // [rec_sum, vq_sum]

  auto im2col = [&](const float* X, int Cin, int Lin, int Lout, int kw, int st,
                    int pad, int Kpad, int relu) {
    long long total = (long long)128 * Lout * Kpad;
    int blocks = (int)((total + 255) / 256);
    k_im2col<<<blocks, 256, 0, stream>>>(X, IM, Cin, Lin, Lout, kw, st, pad, Kpad, relu);
  };
  auto prepw = [&](const float* W, int Cout, int Cin, int kw, int Kpad, int Mpad) {
    int t = Mpad * Kpad;
    k_prep_w<<<(t + 255) / 256, 256, 0, stream>>>(W, WB, Cout, Cin, kw, Kpad, Mpad);
  };
  auto prepwt = [&](const float* W, int Cout, int Cin, int Kpad, int Mpad,
                    int t0, int t1) {
    int t = Mpad * Kpad;
    k_prep_wt<<<(t + 255) / 256, 256, 0, stream>>>(W, WB, Cout, Cin, 4, Kpad, Mpad, t0, t1);
  };
  auto gemm = [&](const float* bias, const float* resid, float* out, int Mout,
                  int Mpad, int Kpad, int Lout, int Lstore, int lsc, int lof,
                  int relu) {
    dim3 grid((128 * Lout) / 256, Mpad / 32);
    k_gemm<<<grid, 256, 0, stream>>>(WB, IM, bias, resid, out, Mout, Kpad,
                                     Lout, Lstore, lsc, lof, relu);
  };

  k_init<<<1, 1, 0, stream>>>(ACC);

  // ---- encoder ----
  prepw(enc1_w, 64, 1, 4, 32, 64);
  im2col(img, 1, 2112, 1056, 4, 2, 1, 32, 0);
  gemm(enc1_b, nullptr, F0, 64, 64, 32, 1056, 1056, 1, 0, 1);

  prepw(enc2_w, 128, 64, 4, 256, 128);
  im2col(F0, 64, 1056, 528, 4, 2, 1, 256, 0);
  gemm(enc2_b, nullptr, F1, 128, 128, 256, 528, 528, 1, 0, 1);

  prepw(enc3_w, 128, 128, 3, 384, 128);
  im2col(F1, 128, 528, 528, 3, 1, 1, 384, 0);
  gemm(enc3_b, nullptr, F0, 128, 128, 384, 528, 528, 1, 0, 0);

  for (int i = 0; i < 4; ++i) {
    prepw(enc_w3 + (long long)i * 128 * 128 * 3, 128, 128, 3, 384, 128);
    im2col(F0, 128, 528, 528, 3, 1, 1, 384, 1);
    gemm(nullptr, nullptr, F1, 128, 128, 384, 528, 528, 1, 0, 0);
    prepw(enc_w1 + (long long)i * 128 * 128, 128, 128, 1, 128, 128);
    im2col(F1, 128, 528, 528, 1, 1, 0, 128, 1);
    gemm(nullptr, F0, F0, 128, 128, 128, 528, 528, 1, 0, 0);
  }

  // pre-VQ (final res-stack ReLU fused into im2col)
  prepw(prevq_w, 32, 128, 1, 128, 32);
  im2col(F0, 128, 528, 528, 1, 1, 0, 128, 1);
  gemm(prevq_b, nullptr, F1, 32, 32, 128, 528, 528, 1, 0, 0);

  // ---- conditional injection + VQ ----
  k_cond<<<16, 256, 0, stream>>>(classes, cp_w, cp_b, CND);
  k_vq<<<67584 / 256, 256, 0, stream>>>(F1, CND, codebook, QB, ACC, 528);

  // ---- decoder ----
  prepw(dec1_w, 128, 32, 3, 96, 128);
  im2col(QB, 32, 528, 528, 3, 1, 1, 96, 0);
  gemm(dec1_b, nullptr, F0, 128, 128, 96, 528, 528, 1, 0, 0);

  for (int i = 0; i < 4; ++i) {
    prepw(dec_w3 + (long long)i * 128 * 128 * 3, 128, 128, 3, 384, 128);
    im2col(F0, 128, 528, 528, 3, 1, 1, 384, 1);
    gemm(nullptr, nullptr, F1, 128, 128, 384, 528, 528, 1, 0, 0);
    prepw(dec_w1 + (long long)i * 128 * 128, 128, 128, 1, 128, 128);
    im2col(F1, 128, 528, 528, 1, 1, 0, 128, 1);
    gemm(nullptr, F0, F0, 128, 128, 128, 528, 528, 1, 0, 0);
  }

  // ConvTranspose1: stride-2 as even/odd parity GEMMs (flipped taps of [Cin][Cout][4])
  // even: out[2j]   = W[..][3]*x[j-1] + W[..][1]*x[j]     (k=2, pad=1)
  // odd : out[2j+1] = W[..][2]*x[j]   + W[..][0]*x[j+1]   (k=2, pad=0)
  prepwt(ct1_w, 64, 128, 256, 64, 3, 1);
  im2col(F0, 128, 528, 528, 2, 1, 1, 256, 1);
  gemm(ct1_b, nullptr, F1, 64, 64, 256, 528, 1056, 2, 0, 1);
  prepwt(ct1_w, 64, 128, 256, 64, 2, 0);
  im2col(F0, 128, 528, 528, 2, 1, 0, 256, 1);
  gemm(ct1_b, nullptr, F1, 64, 64, 256, 528, 1056, 2, 1, 1);

  // ConvTranspose2: Cout=1 (M padded to 32)
  prepwt(ct2_w, 1, 64, 128, 32, 3, 1);
  im2col(F1, 64, 1056, 1056, 2, 1, 1, 128, 0);
  gemm(ct2_b, nullptr, RC, 1, 32, 128, 1056, 2112, 2, 0, 0);
  prepwt(ct2_w, 1, 64, 128, 32, 2, 0);
  im2col(F1, 64, 1056, 1056, 2, 1, 0, 128, 0);
  gemm(ct2_b, nullptr, RC, 1, 32, 128, 1056, 2112, 2, 1, 0);

  // ---- losses ----
  k_recloss<<<256, 256, 0, stream>>>(RC, img, ACC, 270336);
  k_final<<<1, 1, 0, stream>>>(ACC, (float*)d_out);
}
